// Gaussian_Conv2D_5128190951631
// MI455X (gfx1250) — matrix operand order repaired and checked by reading, not compiled, not run on hardware
//
#include <hip/hip_runtime.h>

typedef __attribute__((ext_vector_type(16))) _Float16 v16h;
typedef __attribute__((ext_vector_type(8)))  _Float16 v8h;
typedef __attribute__((ext_vector_type(8)))  float    v8f;

#define BN    16
#define IC    32
#define OC    32
#define HD    256
#define WD    256
#define STRIP 128          // output pixels (w) per workgroup
#define TH    2            // output rows (h) per workgroup
#define ICP   40           // padded ic stride (halves): 80B rows, 16B-aligned
#define NKB   9            // 3x3 taps -> nine K=32 (ic-sweep) blocks
#define S2    (STRIP + 2)  // staged columns incl. halo

__global__ __launch_bounds__(256)
void gaussian_conv2d_wmma(const float* __restrict__ x,
                          const float* __restrict__ wmu,
                          const float* __restrict__ wsig,
                          const float* __restrict__ bmu,
                          const float* __restrict__ bsig,
                          const float* __restrict__ weps,
                          const float* __restrict__ beps,
                          float* __restrict__ out)
{
    // per-sample f16 weights, reordered W[(kh*3+kw)][oc][ic]  (23040 B)
    __shared__ __align__(16) _Float16 Wl[NKB][OC][ICP];
    // f16 input tile: rows h0-1..h0+2, cols wb-1..wb+STRIP, ic innermost (41600 B)
    __shared__ __align__(16) _Float16 Xs[TH + 2][S2][ICP];
    __shared__ float biasS[OC];

    const int t     = threadIdx.x;
    const int wg    = blockIdx.x;             // 16 * 128 * 2 = 4096
    const int strip = wg & 1;
    const int htile = (wg >> 1) & (HD / TH - 1);
    const int b     = wg >> 8;
    const int wb    = strip * STRIP;
    const int h0    = htile * TH;

    // ---- stage per-sample weights: w = mu + eps * |sigma|, cast to f16 ----
    for (int idx = t; idx < OC * IC * 9; idx += 256) {
        const int kw_ = idx % 3;
        const int kh_ = (idx / 3) % 3;
        const int ic  = (idx / 9) % IC;
        const int oc  = idx / (9 * IC);
        const int gi  = ((oc * IC + ic) * 3 + kh_) * 3 + kw_;
        const float v = wmu[gi] + weps[b * (OC * IC * 9) + gi] * fabsf(wsig[gi]);
        Wl[kh_ * 3 + kw_][oc][ic] = (_Float16)v;
    }
    // ---- per-sample bias ----
    if (t < OC)
        biasS[t] = bmu[t] + beps[b * OC + t] * fabsf(bsig[t]);

    // ---- stage input tile (coalesced along columns), zero-padded halo ----
    for (int idx = t; idx < IC * (TH + 2) * S2; idx += 256) {
        const int c  = idx % S2;
        const int r  = (idx / S2) % (TH + 2);
        const int ic = idx / ((TH + 2) * S2);
        const int rg = h0 - 1 + r;
        const int cg = wb - 1 + c;
        float v = 0.0f;
        if ((unsigned)rg < (unsigned)HD && (unsigned)cg < (unsigned)WD)
            v = x[((b * IC + ic) * HD + rg) * WD + cg];
        Xs[r][c][ic] = (_Float16)v;
    }
    __syncthreads();

    const int wave = t >> 5;          // 0..7 -> 16-pixel column tile
    const int lane = t & 31;
    const int n    = lane & 15;       // N column / A row within tile
    const int half = lane >> 4;
    const int p    = wave * 16 + n;   // pixel within strip

    v8f acc00 = {};   // row h0,   OC  0..15
    v8f acc10 = {};   // row h0,   OC 16..31
    v8f acc01 = {};   // row h0+1, OC  0..15
    v8f acc11 = {};   // row h0+1, OC 16..31

    // A-fragment: v16h[j] holds K = (j/8)*16 + half*8 + (j%8)
    const int icA = half * 8;
    // B-fragment: v16h[j] holds K = (j/8)*16 + half*8 + (j%8)
    const int icB = half * 8;

#pragma unroll
    for (int kb = 0; kb < NKB; ++kb) {
        const int kh_ = kb / 3;
        const int kw_ = kb % 3;

        const v8h w00 = *(const v8h*)&Wl[kb][n][icA];
        const v8h w01 = *(const v8h*)&Wl[kb][n][icA + 16];
        const v8h w10 = *(const v8h*)&Wl[kb][n + 16][icA];
        const v8h w11 = *(const v8h*)&Wl[kb][n + 16][icA + 16];
        const v8h x00 = *(const v8h*)&Xs[kh_][p + kw_][icB];
        const v8h x01 = *(const v8h*)&Xs[kh_][p + kw_][icB + 16];
        const v8h x10 = *(const v8h*)&Xs[kh_ + 1][p + kw_][icB];
        const v8h x11 = *(const v8h*)&Xs[kh_ + 1][p + kw_][icB + 16];

        v16h a0, a1, bb0, bb1;
#pragma unroll
        for (int j = 0; j < 8; ++j) {
            a0[j]  = w00[j]; a0[8 + j]  = w01[j];
            a1[j]  = w10[j]; a1[8 + j]  = w11[j];
            bb0[j] = x00[j]; bb0[8 + j] = x01[j];
            bb1[j] = x10[j]; bb1[8 + j] = x11[j];
        }

        acc00 = __builtin_amdgcn_wmma_f32_16x16x32_f16(
                    false, a0, false, bb0, (short)0, acc00, false, false);
        acc10 = __builtin_amdgcn_wmma_f32_16x16x32_f16(
                    false, a1, false, bb0, (short)0, acc10, false, false);
        acc01 = __builtin_amdgcn_wmma_f32_16x16x32_f16(
                    false, a0, false, bb1, (short)0, acc01, false, false);
        acc11 = __builtin_amdgcn_wmma_f32_16x16x32_f16(
                    false, a1, false, bb1, (short)0, acc11, false, false);
    }

    // ---- epilogue: C/D layout -> VGPR r: M = r + 8*half, N = lane%16 ----
    const int wp = wb + p;            // global w coordinate
#pragma unroll
    for (int r = 0; r < 8; ++r) {
        const int oc0 = r + 8 * half;
        const int oc1 = 16 + r + 8 * half;
        const float bi0 = biasS[oc0];
        const float bi1 = biasS[oc1];
        out[((b * OC + oc0) * HD + h0) * WD + wp]     = acc00[r] + bi0;
        out[((b * OC + oc1) * HD + h0) * WD + wp]     = acc10[r] + bi1;
        out[((b * OC + oc0) * HD + h0 + 1) * WD + wp] = acc01[r] + bi0;
        out[((b * OC + oc1) * HD + h0 + 1) * WD + wp] = acc11[r] + bi1;
    }
}

extern "C" void kernel_launch(void* const* d_in, const int* in_sizes, int n_in,
                              void* d_out, int out_size, void* d_ws, size_t ws_size,
                              hipStream_t stream) {
    (void)in_sizes; (void)n_in; (void)out_size; (void)d_ws; (void)ws_size;
    const float* x    = (const float*)d_in[0];
    const float* wmu  = (const float*)d_in[1];
    const float* wsig = (const float*)d_in[2];
    const float* bmu  = (const float*)d_in[3];
    const float* bsig = (const float*)d_in[4];
    const float* weps = (const float*)d_in[5];
    const float* beps = (const float*)d_in[6];
    float* out = (float*)d_out;

    const int nblocks = BN * (HD / TH) * (WD / STRIP);   // 4096
    gaussian_conv2d_wmma<<<dim3(nblocks), dim3(256), 0, stream>>>(
        x, wmu, wsig, bmu, bsig, weps, beps, out);
}
